// ginarCell_87041807221035
// MI455X (gfx1250) — matrix operand order repaired and checked by reading, not compiled, not run on hardware
//
#include <hip/hip_runtime.h>
#include <hip/hip_bf16.h>

#define BB 16
#define NN 1024
#define CIN 3
#define EE 64
#define GDIM 16
#define E2 128

typedef __attribute__((ext_vector_type(16))) _Float16 v16h;
typedef __attribute__((ext_vector_type(8)))  _Float16 v8h;
typedef __attribute__((ext_vector_type(8)))  float    v8f;

__device__ __forceinline__ float lrelu(float v){ return v >= 0.f ? v : 0.01f*v; }

// A fragment: halves 0..7 = K+0..7, halves 8..15 = K+16..23 (per lane group)
__device__ __forceinline__ v16h ldfrag_a(const _Float16* p){
  v8h lo = *(const v8h*)(p);
  v8h hi = *(const v8h*)(p + 16);
  return __builtin_shufflevector(lo, hi, 0,1,2,3,4,5,6,7,8,9,10,11,12,13,14,15);
}
// B fragment: halves 0..7 = K+0..7, halves 8..15 = K+16..23 (per lane group)
__device__ __forceinline__ v16h ldfrag_b(const _Float16* p){
  v8h lo = *(const v8h*)(p);
  v8h hi = *(const v8h*)(p + 16);
  return __builtin_shufflevector(lo, hi, 0,1,2,3,4,5,6,7,8,9,10,11,12,13,14,15);
}

// ---------------- K1: h0[b,e,n] = lrelu(emb_w @ x + emb_b), stored f16 ----------------
__global__ void k_emb(const float* __restrict__ x, const float* __restrict__ w,
                      const float* __restrict__ bias, _Float16* __restrict__ h0){
  int idx = blockIdx.x*blockDim.x + threadIdx.x;
  if (idx >= BB*EE*NN) return;
  int n = idx % NN;
  int e = (idx / NN) % EE;
  int b = idx / (NN*EE);
  float acc = bias[e];
  #pragma unroll
  for (int c=0;c<CIN;c++) acc += w[e*CIN+c]*x[(b*CIN+c)*NN + n];
  h0[idx] = (_Float16)lrelu(acc);
}

// ---------------- generic f32 -> f16 conversion (for weight inputs) ----------------
__global__ void k_cvt16(const float* __restrict__ in, _Float16* __restrict__ out, long n){
  long i = (long)blockIdx.x*blockDim.x + threadIdx.x;
  if (i < n) out[i] = (_Float16)in[i];
}

// ------------- WMMA GEMM: C[M,N] (+)= A[M,K] * B[N,K]^T, f16 operands, f32 accum -------------
// 16(M) x 64(N) register tile per wave32; A fragment reused across 4 WMMAs;
// software double-buffering: k+32 fragments prefetched before the 4-WMMA burst.
// blockDim=(32,4); grid = (N/64, ceil(M/64), batches).
__global__ void k_wmma_abt(const _Float16* __restrict__ A, long sA,
                           const _Float16* __restrict__ Bm, long sB,
                           float* __restrict__ Cm, long sC,
                           int M, int N, int K, int acc){
  A  += (long)blockIdx.z * sA;
  Bm += (long)blockIdx.z * sB;
  Cm += (long)blockIdx.z * sC;
  int tileN4 = blockIdx.x;                         // 64-column group
  int tileM  = blockIdx.y*blockDim.y + threadIdx.y;
  if (tileM*16 >= M) return;                       // wave-uniform: EXEC stays full
  int lane = threadIdx.x;
  int l15 = lane & 15, hi = lane >> 4;
  long arow  = tileM*16 + l15;                     // A fragment: lane = row
  long bcol0 = (long)tileN4*64 + l15;              // B fragment: lane = col
  int aKoff = hi*8;                                // A half-split offset
  int bKoff = hi*8;                                // B half-split offset
  int crow  = tileM*16 + hi*8;
  int ccol0 = tileN4*64 + l15;

  const _Float16* aB = A  + arow*K  + aKoff;       // loop-invariant bases
  const _Float16* bB = Bm + bcol0*K + bKoff;
  const long bs = (long)16*K;                      // stride between column groups

  v8f cacc[4] = {{},{},{},{}};
  if (acc){
    #pragma unroll
    for (int t=0;t<4;t++)
      #pragma unroll
      for (int r=0;r<8;r++) cacc[t][r] = Cm[(long)(crow+r)*N + ccol0 + t*16];
  }

  // prologue: fragments for k = 0
  v16h av = ldfrag_a(aB);
  v16h b0 = ldfrag_b(bB);
  v16h b1 = ldfrag_b(bB + bs);
  v16h b2 = ldfrag_b(bB + 2*bs);
  v16h b3 = ldfrag_b(bB + 3*bs);

  for (int k0=32; k0<K; k0+=32){
    // prefetch next step (independent of the WMMAs below -> overlaps latency)
    v16h avn = ldfrag_a(aB + k0);
    v16h b0n = ldfrag_b(bB + k0);
    v16h b1n = ldfrag_b(bB + bs   + k0);
    v16h b2n = ldfrag_b(bB + 2*bs + k0);
    v16h b3n = ldfrag_b(bB + 3*bs + k0);
    cacc[0] = __builtin_amdgcn_wmma_f32_16x16x32_f16(false, av, false, b0, (short)0, cacc[0], false, false);
    cacc[1] = __builtin_amdgcn_wmma_f32_16x16x32_f16(false, av, false, b1, (short)0, cacc[1], false, false);
    cacc[2] = __builtin_amdgcn_wmma_f32_16x16x32_f16(false, av, false, b2, (short)0, cacc[2], false, false);
    cacc[3] = __builtin_amdgcn_wmma_f32_16x16x32_f16(false, av, false, b3, (short)0, cacc[3], false, false);
    av = avn; b0 = b0n; b1 = b1n; b2 = b2n; b3 = b3n;
  }
  cacc[0] = __builtin_amdgcn_wmma_f32_16x16x32_f16(false, av, false, b0, (short)0, cacc[0], false, false);
  cacc[1] = __builtin_amdgcn_wmma_f32_16x16x32_f16(false, av, false, b1, (short)0, cacc[1], false, false);
  cacc[2] = __builtin_amdgcn_wmma_f32_16x16x32_f16(false, av, false, b2, (short)0, cacc[2], false, false);
  cacc[3] = __builtin_amdgcn_wmma_f32_16x16x32_f16(false, av, false, b3, (short)0, cacc[3], false, false);

  #pragma unroll
  for (int t=0;t<4;t++)
    #pragma unroll
    for (int r=0;r<8;r++) Cm[(long)(crow+r)*N + ccol0 + t*16] = cacc[t][r];
}

// ------ K3: h = (raw + emb2_b[n]) @ att_W ; e1 = h.a1 ; e2 = h.a2 ------
__global__ void k_h_e12(const float* __restrict__ raw, const float* __restrict__ emb2b,
                        const float* __restrict__ attW, const float* __restrict__ atta,
                        float* __restrict__ h, float* __restrict__ e1, float* __restrict__ e2){
  int bn = blockIdx.x;            // b*N + n
  int t  = threadIdx.x;           // 0..63 (output channel e')
  int n  = bn & (NN-1);
  __shared__ float row[EE];
  __shared__ float red[EE];
  row[t] = raw[(long)bn*EE + t] + emb2b[n];
  __syncthreads();
  float acc = 0.f;
  #pragma unroll 8
  for (int c=0;c<EE;c++) acc += row[c]*attW[c*EE + t];
  h[(long)bn*EE + t] = acc;
  red[t] = acc * atta[t];
  __syncthreads();
  for (int s=32;s>0;s>>=1){ if (t<s) red[t]+=red[t+s]; __syncthreads(); }
  if (t==0) e1[bn]=red[0];
  __syncthreads();
  red[t] = acc * atta[EE+t];
  __syncthreads();
  for (int s=32;s>0;s>>=1){ if (t<s) red[t]+=red[t+s]; __syncthreads(); }
  if (t==0) e2[bn]=red[0];
}

__global__ void k_maxe2(const float* __restrict__ e2, float* __restrict__ mx){
  int b = blockIdx.x; int t = threadIdx.x;
  __shared__ float sh[256];
  float m = -1e30f;
  for (int j=t;j<NN;j+=256) m = fmaxf(m, e2[b*NN+j]);
  sh[t]=m; __syncthreads();
  for (int s=128;s>0;s>>=1){ if (t<s) sh[t]=fmaxf(sh[t],sh[t+s]); __syncthreads(); }
  if (t==0) mx[b]=sh[0];
}

// ---- K4: streaming row softmax attention (mask is provably all-pass) ----
__global__ void k_attn(const float* __restrict__ h, const float* __restrict__ e1,
                       const float* __restrict__ e2, const float* __restrict__ mxe2,
                       float* __restrict__ xa, float* __restrict__ xat){
  int bi = blockIdx.x;
  int b = bi / NN, i = bi & (NN-1);
  int t = threadIdx.x;            // 0..63 channel
  float E1 = e1[bi];
  float rm = lrelu(E1 + mxe2[b]); // lrelu monotone => exact rowmax
  float den=0.f, acc=0.f;
  const float* hb  = h  + (long)b*NN*EE;
  const float* e2b = e2 + b*NN;
  for (int j=0;j<NN;j++){
    float w = __expf(lrelu(E1 + e2b[j]) - rm);
    den += w;
    acc += w * hb[j*EE + t];
  }
  float v = acc/den; v = v>0.f ? v : 0.f;     // relu(att_sc @ h)
  xa [((long)b*EE + t)*NN + i] = v;           // [B,E,N] f32
  xat[(long)bi*EE + t]         = v;           // [B,N,E] f32
}

// ---- K5: normalized-laplacian, stored pre-transposed f16: gt[j,i] = g[i,j] ----
__global__ void k_deg(const float* __restrict__ g, float* __restrict__ d){
  int row = blockIdx.x;           // 0..2N-1
  int k = row / NN, i = row % NN;
  int t = threadIdx.x;
  __shared__ float sh[256];
  const float* gr = g + ((long)k*NN + i)*NN;
  float s=0.f;
  for (int j=t;j<NN;j+=256) s += gr[j] + ((j==i)?1.f:0.f);
  sh[t]=s; __syncthreads();
  for (int st=128;st>0;st>>=1){ if (t<st) sh[t]+=sh[t+st]; __syncthreads(); }
  if (t==0){ float sum = sh[0]; d[row] = (sum>0.f)? rsqrtf(sum) : 0.f; }
}

__global__ void k_lapt(const float* __restrict__ g, const float* __restrict__ d,
                       _Float16* __restrict__ g1t, _Float16* __restrict__ g2t){
  long idx = (long)blockIdx.x*blockDim.x + threadIdx.x;
  if (idx >= 2L*NN*NN) return;
  int  k   = (int)(idx / ((long)NN*NN));
  long rem = idx - (long)k*NN*NN;
  int a = (int)(rem / NN), bcol = (int)(rem % NN);
  float m = g[idx] + ((a==bcol)?1.f:0.f);
  float v = d[k*NN + bcol] * m * d[k*NN + a]; // gt[a,b] = d[b]*m[a,b]*d[a]
  (k==0 ? g1t : g2t)[rem] = (_Float16)v;
}

__global__ void k_gle(const float* __restrict__ GL, const float* __restrict__ W,
                      float* __restrict__ gle){
  int idx = blockIdx.x*blockDim.x + threadIdx.x;
  if (idx>=NN*EE) return;
  int e = idx % EE, n = idx / EE;
  float acc=0.f;
  #pragma unroll
  for (int g=0;g<GDIM;g++) acc += GL[n*GDIM+g]*W[e*GDIM+g];
  gle[idx]=acc;
}

// ---- K7: ge[b,i,k] = concat(xat[b,i,:], gl_e[i,:]) @ GLlin2_w^T, stored f16 ----
__global__ void k_ge(const float* __restrict__ xat, const float* __restrict__ gle,
                     const float* __restrict__ W2, _Float16* __restrict__ ge){
  int bi = blockIdx.x; int t = threadIdx.x;   // 0..127
  int i = bi & (NN-1);
  __shared__ float catv[E2];
  catv[t] = (t < EE) ? xat[(long)bi*EE + t] : gle[i*EE + (t-EE)];
  __syncthreads();
  float acc=0.f;
  for (int c=0;c<E2;c++) acc += catv[c]*W2[t*E2+c];
  ge[(long)bi*E2 + t] = (_Float16)acc;
}

// ---- 1x1 conv over channels (K=64, VALU), f32 in -> f16 out (feeds WMMA) ----
__global__ void k_conv1x1(const float* __restrict__ in, const float* __restrict__ W,
                          const float* __restrict__ bias, _Float16* __restrict__ out, int nb){
  long idx = (long)blockIdx.x*blockDim.x + threadIdx.x;
  if (idx >= (long)nb*EE*NN) return;
  int n = (int)(idx % NN);
  int o = (int)((idx/NN) % EE);
  long b = idx/((long)NN*EE);
  const float* ib = in + b*EE*NN;
  float acc = bias ? bias[o] : 0.f;
  #pragma unroll 8
  for (int c=0;c<EE;c++) acc += W[o*EE+c]*ib[c*NN+n];
  out[idx]=(_Float16)acc;
}

// ---- graph_learn: glt[j,i] = softmax_j(relu(gg[i,:])) + eye, transposed, f16 ----
__global__ void k_smax_t(const float* __restrict__ gg, _Float16* __restrict__ glt){
  int i = blockIdx.x; int t = threadIdx.x;
  __shared__ float sh[256];
  const float* r = gg + (long)i*NN;
  float mx = 0.f;
  for (int j=t;j<NN;j+=256){ float v=r[j]; v=v>0?v:0; mx=fmaxf(mx,v); }
  sh[t]=mx; __syncthreads();
  for (int s=128;s>0;s>>=1){ if(t<s) sh[t]=fmaxf(sh[t],sh[t+s]); __syncthreads(); }
  float Mv = sh[0]; __syncthreads();
  float sum=0.f;
  for (int j=t;j<NN;j+=256){ float v=r[j]; v=v>0?v:0; sum += __expf(v-Mv); }
  sh[t]=sum; __syncthreads();
  for (int s=128;s>0;s>>=1){ if(t<s) sh[t]+=sh[t+s]; __syncthreads(); }
  float S = sh[0];
  for (int j=t;j<NN;j+=256){
    float v=r[j]; v=v>0?v:0;
    glt[(long)j*NN + i] = (_Float16)(__expf(v-Mv)/S + ((i==j)?1.f:0.f));
  }
}

// ---- LayerNorm stats over [E,N] per batch, both branches at once ----
__global__ void k_stats(const float* __restrict__ xs, const float* __restrict__ gs,
                        float* __restrict__ st){
  int b = blockIdx.x; int t = threadIdx.x;
  __shared__ float sh[256];
  const float* xb = xs + (long)b*EE*NN;
  const float* gb = gs + (long)b*EE*NN;
  float s1=0,s2=0,s3=0,s4=0;
  for (int k=t;k<EE*NN;k+=256){
    float v=xb[k]; s1+=v; s2+=v*v;
    float u=gb[k]; s3+=u; s4+=u*u;
  }
  float in4[4]={s1,s2,s3,s4}; float res[4];
  for (int q=0;q<4;q++){
    __syncthreads(); sh[t]=in4[q]; __syncthreads();
    for (int s=128;s>0;s>>=1){ if(t<s) sh[t]+=sh[t+s]; __syncthreads(); }
    res[q]=sh[0];
  }
  if (t==0){
    float inv = 1.f/(float)(EE*NN);
    float mx = res[0]*inv, vx = res[1]*inv - mx*mx;
    float mg = res[2]*inv, vg = res[3]*inv - mg*mg;
    st[b*4+0]=mx; st[b*4+1]=rsqrtf(vx+1e-5f);
    st[b*4+2]=mg; st[b*4+3]=rsqrtf(vg+1e-5f);
  }
}

// ---- fused epilogue: LN affine, exact gelu, gates (ft==rt==g), elu ----
__global__ void k_final(const float* __restrict__ xs, const float* __restrict__ gs,
                        const float* __restrict__ ct, const float* __restrict__ xa,
                        const float* __restrict__ lnw, const float* __restrict__ lnb,
                        const float* __restrict__ st, float* __restrict__ out){
  long idx = (long)blockIdx.x*blockDim.x + threadIdx.x;
  if (idx >= (long)BB*EE*NN) return;
  int en = (int)(idx % ((long)EE*NN));
  int b  = (int)(idx / ((long)EE*NN));
  float xn = (xs[idx]-st[b*4+0])*st[b*4+1]*lnw[en] + lnb[en];
  float gp = (gs[idx]-st[b*4+2])*st[b*4+3]*lnw[en] + lnb[en];
  float g  = 0.5f*gp*(1.0f + erff(gp*0.70710678118654752f)); // exact gelu
  float ctn = g*ct[idx] + xn - g*xn;
  float el  = ctn>0.f ? ctn : (__expf(ctn)-1.f);             // elu
  float ht  = g*el + xa[idx] - g*xa[idx];
  out[idx] = ht;
  out[(long)BB*EE*NN + idx] = ctn;
}

extern "C" void kernel_launch(void* const* d_in, const int* in_sizes, int n_in,
                              void* d_out, int out_size, void* d_ws, size_t ws_size,
                              hipStream_t stream){
  const float* x      =(const float*)d_in[0];
  const float* ct     =(const float*)d_in[1];
  const float* graph  =(const float*)d_in[2];
  const float* emb_w  =(const float*)d_in[3];
  const float* emb_b  =(const float*)d_in[4];
  const float* emb2_w =(const float*)d_in[5];
  const float* emb2_b =(const float*)d_in[6];
  const float* att_W  =(const float*)d_in[7];
  const float* att_a  =(const float*)d_in[8];
  // d_in[9] = att_GL: unused — softmax(.)+I is strictly positive, mask is a no-op
  const float* lin1_w =(const float*)d_in[10];
  const float* lin2_w =(const float*)d_in[11];
  const float* lin2_b =(const float*)d_in[12];
  const float* ln_w   =(const float*)d_in[13];
  const float* ln_b   =(const float*)d_in[14];
  const float* GLp    =(const float*)d_in[15];
  const float* GLlinw =(const float*)d_in[16];
  const float* GLlin2 =(const float*)d_in[17];
  float* out = (float*)d_out;
  (void)in_sizes; (void)n_in; (void)out_size; (void)ws_size;

  char* base = (char*)d_ws;
  size_t off = 0;
  auto allocB = [&](size_t bytes)->void*{
    void* p = base + off;
    off = (off + bytes + 255) & ~(size_t)255;
    return p;
  };
  const size_t BEN = (size_t)BB*EE*NN;
  _Float16* h0    = (_Float16*)allocB(BEN*2);          // [B,E,N] f16
  _Float16* emb2h = (_Float16*)allocB((size_t)NN*NN*2);
  float*    raw   = (float*)   allocB(BEN*4);          // inp pre-bias, [B,N,E]
  float*    h     = (float*)   allocB(BEN*4);          // [B,N,E]
  float*    e1    = (float*)   allocB((size_t)BB*NN*4);
  float*    e2    = (float*)   allocB((size_t)BB*NN*4);
  float*    mxe2  = (float*)   allocB(BB*4);
  float*    xa    = (float*)   allocB(BEN*4);          // [B,E,N]
  float*    xat   = (float*)   allocB(BEN*4);          // [B,N,E]
  float*    dvec  = (float*)   allocB(2*NN*4);
  _Float16* g1t   = (_Float16*)allocB((size_t)NN*NN*2);
  _Float16* g2t   = (_Float16*)allocB((size_t)NN*NN*2);
  float*    gle   = (float*)   allocB((size_t)NN*EE*4);
  _Float16* ge    = (_Float16*)allocB((size_t)BB*NN*E2*2);
  _Float16* l1x   = (_Float16*)allocB(BEN*2);
  _Float16* l2x   = (_Float16*)allocB(BEN*2);
  float*    xsum  = (float*)   allocB(BEN*4);
  float*    gsum  = (float*)   allocB(BEN*4);
  float*    gg    = (float*)   allocB((size_t)NN*NN*4);   // per-batch reuse
  _Float16* glt   = (_Float16*)allocB((size_t)NN*NN*2);   // per-batch reuse
  float*    t1    = (float*)   allocB((size_t)EE*NN*4);
  _Float16* u1    = (_Float16*)allocB((size_t)EE*NN*2);
  float*    stats = (float*)   allocB(64*4);

  dim3 wblk(32,4);

  k_emb<<<(int)((BEN+255)/256),256,0,stream>>>(x, emb_w, emb_b, h0);
  k_cvt16<<<(int)(((long)NN*NN+255)/256),256,0,stream>>>(emb2_w, emb2h, (long)NN*NN);
  // raw[b][m,e] = emb2_w[m,:] · h0[b][e,:]   (M=1024 rows, 64 cols, K=1024)
  k_wmma_abt<<<dim3(EE/64,NN/64,BB),wblk,0,stream>>>(
      emb2h, 0, h0, (long)EE*NN, raw, (long)NN*EE, NN, EE, NN, 0);
  k_h_e12<<<BB*NN,EE,0,stream>>>(raw, emb2_b, att_W, att_a, h, e1, e2);
  k_maxe2<<<BB,256,0,stream>>>(e2, mxe2);
  k_attn<<<BB*NN,EE,0,stream>>>(h, e1, e2, mxe2, xa, xat);
  k_deg<<<2*NN,256,0,stream>>>(graph, dvec);
  k_lapt<<<(int)((2L*NN*NN+255)/256),256,0,stream>>>(graph, dvec, g1t, g2t);
  k_gle<<<(NN*EE+255)/256,256,0,stream>>>(GLp, GLlinw, gle);
  k_ge<<<BB*NN,E2,0,stream>>>(xat, gle, GLlin2, ge);
  k_conv1x1<<<(int)((BEN+255)/256),256,0,stream>>>(xa, lin1_w, nullptr, l1x, BB);
  k_conv1x1<<<(int)((BEN+255)/256),256,0,stream>>>(xa, lin2_w, lin2_b,  l2x, BB);

  for (int b=0;b<BB;b++){
    const _Float16* geb = ge  + (size_t)b*NN*E2;
    _Float16* l1b = l1x  + (size_t)b*EE*NN;
    _Float16* l2b = l2x  + (size_t)b*EE*NN;
    float* xsb = xsum + (size_t)b*EE*NN;
    float* gsb = gsum + (size_t)b*EE*NN;
    // gg = ge·geᵀ  (1024x1024, K=128)
    k_wmma_abt<<<dim3(NN/64,NN/64,1),wblk,0,stream>>>(geb,0, geb,0, gg,0, NN,NN,E2, 0);
    k_smax_t<<<NN,256,0,stream>>>(gg, glt);     // graph_learnᵀ (eye + softmax), f16
    // x branch: xsum = l1x·gl ... += lin1(l1x·g1)·g2
    k_wmma_abt<<<dim3(NN/64,1,1),wblk,0,stream>>>(l1b,0, glt,0, xsb,0, EE,NN,NN, 0);
    k_wmma_abt<<<dim3(NN/64,1,1),wblk,0,stream>>>(l1b,0, g1t,0, t1 ,0, EE,NN,NN, 0);
    k_conv1x1<<<(EE*NN+255)/256,256,0,stream>>>(t1, lin1_w, nullptr, u1, 1);
    k_wmma_abt<<<dim3(NN/64,1,1),wblk,0,stream>>>(u1 ,0, g2t,0, xsb,0, EE,NN,NN, 1);
    // g branch: gsum = l2x·gl ... += lin2(l2x·g1)·g2
    k_wmma_abt<<<dim3(NN/64,1,1),wblk,0,stream>>>(l2b,0, glt,0, gsb,0, EE,NN,NN, 0);
    k_wmma_abt<<<dim3(NN/64,1,1),wblk,0,stream>>>(l2b,0, g1t,0, t1 ,0, EE,NN,NN, 0);
    k_conv1x1<<<(EE*NN+255)/256,256,0,stream>>>(t1, lin2_w, lin2_b, u1, 1);
    k_wmma_abt<<<dim3(NN/64,1,1),wblk,0,stream>>>(u1 ,0, g2t,0, gsb,0, EE,NN,NN, 1);
  }

  k_stats<<<BB,256,0,stream>>>(xsum, gsum, stats);
  k_final<<<(int)((BEN+255)/256),256,0,stream>>>(xsum, gsum, ct, xa, ln_w, ln_b, stats, out);
}
